// HoughVoting_66743791780312
// MI455X (gfx1250) — matrix operand order repaired and checked by reading, not compiled, not run on hardware
//
#include <hip/hip_runtime.h>
#include <hip/hip_bf16.h>
#include <math.h>

// ---------------- constants matching the reference ----------------
#define NUM_CLASSES 10
#define HH 480
#define WW 640
#define SKIP 8
#define GX 80              // W / SKIP
#define GY 60              // H / SKIP
#define P_PTS 4800         // GX * GY, divisible by 32
#define NTILES 300         // P_PTS / 16 center tiles
#define INLIER_TH 0.9f
#define EPS_F 1e-6f

typedef __attribute__((ext_vector_type(16))) _Float16 v16h;
typedef __attribute__((ext_vector_type(8)))  float    v8f;

// ---------------------------------------------------------------------------
// Pass 1: gather per-sampled-pixel features into SoA arrays.
//   u,v : normalized direction, PRE-MASKED by (lab>0). With u=v=0 the pair
//         numerator is 0 and can never beat 0.9*rd (rd >= eps > 0), so the
//         (lab>0) term of the inlier predicate is folded into the data.
//   d   : exp(clip(z,-3,3));  lab : class id (kept for counts / one-hot)
// ---------------------------------------------------------------------------
__global__ __launch_bounds__(256)
void hough_prep_kernel(const int* __restrict__ label,
                       const float* __restrict__ vp,
                       float* __restrict__ u_arr, float* __restrict__ v_arr,
                       float* __restrict__ d_arr, int* __restrict__ lab_arr)
{
    int i = blockIdx.x * blockDim.x + threadIdx.x;
    if (i >= P_PTS) return;
    int gy = i / GX;
    int gx = i - gy * GX;
    int pix = (gy * SKIP) * WW + (gx * SKIP);
    int lb = label[pix];
    const float* base = vp + (long)pix * (3 * NUM_CLASSES) + 3 * lb;
    float uu = base[0];
    float vv = base[1];
    float zz = fminf(fmaxf(base[2], -3.0f), 3.0f);
    float nrm = sqrtf(uu * uu + vv * vv) + EPS_F;
    float msk = (lb > 0) ? 1.0f : 0.0f;
    u_arr[i] = msk * uu / nrm;
    v_arr[i] = msk * vv / nrm;
    d_arr[i] = expf(zz);
    lab_arr[i] = lb;
}

// ---------------------------------------------------------------------------
// Pass 2: voting. One wave per tile of 16 candidate centers.
// Per 32-pixel chunk:
//   * each lane stages its pixel's {u,v,x,y} (float4) and {d,lab} (float2)
//   * each lane then builds the B-tile elements IT owns in this kernel's contiguous-K
//     layout (N = lane%16, K = h + 16*(lane/16)) directly from LDS float4s,
//     and the A-tile one-hot rows (M = lane%16) from the {d,lab} pairs
//   * two v_wmma_f32_16x16x32_f16 accumulate hough[16,16] and dsum[16,16]
// Predicate per pair: u*dx + v*dy > 0.9 * (sqrt(dx^2+dy^2) + eps)  (no div).
// ---------------------------------------------------------------------------
__global__ __launch_bounds__(32)
void hough_vote_kernel(const float* __restrict__ u_arr,
                       const float* __restrict__ v_arr,
                       const float* __restrict__ d_arr,
                       const int* __restrict__ lab_arr,
                       float* __restrict__ hough,   // [NUM_CLASSES][P_PTS]
                       float* __restrict__ dsum)    // [NUM_CLASSES][P_PTS]
{
    __shared__ float4 sUVXY[32];   // {u, v, px, py} per pixel in chunk
    __shared__ float2 sDL[32];     // {d, (float)lab} per pixel in chunk

    const int lane  = threadIdx.x;      // 0..31
    const int group = lane >> 4;        // 0 or 1
    const int nloc  = lane & 15;        // center column / class row
    const int n0    = blockIdx.x * 16;

    // this lane's candidate-center coordinates (fixed for the whole kernel)
    const int cidx = n0 + nloc;
    const int cgy  = cidx / GX;
    const int cgx  = cidx - cgy * GX;
    const float ccx = (float)(cgx * SKIP);
    const float ccy = (float)(cgy * SKIP);
    const float labf = (float)nloc;

    v8f accH = {};
    v8f accD = {};

    for (int p0 = 0; p0 < P_PTS; p0 += 32) {
        const int i = p0 + lane;
        const int gy = i / GX;
        const int gx = i - gy * GX;
        sUVXY[lane] = make_float4(u_arr[i], v_arr[i],
                                  (float)(gx * SKIP), (float)(gy * SKIP));
        sDL[lane]   = make_float2(d_arr[i], (float)lab_arr[i]);
        __syncthreads();

        // B (32x16, f16): lane owns column nloc; half h -> K = h + 16*group
        v16h b;
#pragma unroll
        for (int h = 0; h < 16; ++h) {
            const float4 q = sUVXY[h + 16 * group];
            const float dx = ccx - q.z;
            const float dy = ccy - q.w;
            const float rd = sqrtf(dx * dx + dy * dy) + EPS_F;
            const float num = q.x * dx + q.y * dy;
            b[h] = (_Float16)((num > INLIER_TH * rd) ? 1.0f : 0.0f);
        }

        // A (16x32, f16): lane row M = nloc; half h -> K = h + 16*group, as B
        v16h a1, a2;
#pragma unroll
        for (int h = 0; h < 16; ++h) {
            const int kA = h + 16 * group;
            const float2 dl = sDL[kA];
            const bool hit = (dl.y == labf);
            a1[h] = (_Float16)(hit ? 1.0f : 0.0f);
            a2[h] = (_Float16)(hit ? dl.x : 0.0f);
        }

        // D = A x B + C (f32 accumulate); EXEC all-ones here.
        accH = __builtin_amdgcn_wmma_f32_16x16x32_f16(
            false, a1, false, b, (short)0, accH, false, false);
        accD = __builtin_amdgcn_wmma_f32_16x16x32_f16(
            false, a2, false, b, (short)0, accD, false, false);
        __syncthreads();
    }

    // C/D layout: VGPR j, lane L -> M = j + 8*(L>=16), N = L%16
#pragma unroll
    for (int j = 0; j < 8; ++j) {
        const int m = j + 8 * group;
        if (m < NUM_CLASSES) {
            hough[m * P_PTS + n0 + nloc] = accH[j];
            dsum [m * P_PTS + n0 + nloc] = accD[j];
        }
    }
}

// ---------------------------------------------------------------------------
// Pass 3: per-class argmax (first-index tie-break), counts, box/pose math,
// and zero-fill of the train-only output slices. One block per class.
// Output layout (concat, flat):
//   [0,63)    top_box   [9][7]
//   [63,126)  top_pose  [9][7]
//   [126,486) top_target[9][40]  (zeros)
//   [486,846) top_weight[9][40]  (zeros)
//   [846,855) top_domain[9]      (zeros; batch 0)
// ---------------------------------------------------------------------------
__global__ __launch_bounds__(256)
void hough_finalize_kernel(const float* __restrict__ hough,
                           const float* __restrict__ dsum,
                           const int* __restrict__ lab_arr,
                           const float* __restrict__ extents, // [10][3]
                           const float* __restrict__ meta,    // [9]
                           float* __restrict__ out)
{
    __shared__ float sv[256];
    __shared__ int   si[256];
    __shared__ float sc[256];

    const int k = blockIdx.x;     // class 0..9
    const int t = threadIdx.x;

    float bestv = -1.0f;
    int   besti = 0;
    float cnt   = 0.0f;
    for (int c = t; c < P_PTS; c += 256) {
        const float v = hough[k * P_PTS + c];
        if (v > bestv) { bestv = v; besti = c; }   // c ascending -> first max
        cnt += (lab_arr[c] == k) ? 1.0f : 0.0f;
    }
    sv[t] = bestv; si[t] = besti; sc[t] = cnt;
    __syncthreads();
    for (int s = 128; s > 0; s >>= 1) {
        if (t < s) {
            if (sv[t + s] > sv[t] ||
                (sv[t + s] == sv[t] && si[t + s] < si[t])) {
                sv[t] = sv[t + s]; si[t] = si[t + s];
            }
            sc[t] += sc[t + s];
        }
        __syncthreads();
    }

    if (k >= 1) {
        const int r = k - 1;
        // zero train-only outputs
        if (t < 40) {
            out[126 + r * 40 + t] = 0.0f;
            out[486 + r * 40 + t] = 0.0f;
        }
        if (t == 1) out[846 + r] = 0.0f;   // domain (batch 0)

        if (t == 0) {
            const float votes  = sv[0];
            const int   best   = si[0];
            const float counts = sc[0];
            const float ds   = dsum[k * P_PTS + best];
            const float davg = ds / fmaxf(votes, 1.0f);
            const int bgy = best / GX;
            const int bgx = best - bgy * GX;
            const float cx = (float)(bgx * SKIP);
            const float cy = (float)(bgy * SKIP);
            const bool valid = (votes > 20.0f) &&
                               (votes > 0.1f * counts) &&
                               (counts * (float)(SKIP * SKIP) > 500.0f);
            const float fx  = meta[0], ppx = meta[2];
            const float fy  = meta[4], ppy = meta[5];
            const float ex = extents[k * 3 + 0];
            const float ey = extents[k * 3 + 1];
            const float ez = extents[k * 3 + 2];
            const float diam = sqrtf(ex * ex + ey * ey + ez * ez) + EPS_F;
            const float hx = 0.5f * diam * fx / fmaxf(davg, EPS_F);
            const float hy = 0.5f * diam * fy / fmaxf(davg, EPS_F);
            const float score = valid ? votes : 0.0f;

            float* box = out + r * 7;
            box[0] = 0.0f;            // batch index
            box[1] = (float)k;        // class id
            box[2] = cx - hx;
            box[3] = cy - hy;
            box[4] = cx + hx;
            box[5] = cy + hy;
            box[6] = score;

            float* pose = out + 63 + r * 7;
            pose[0] = 1.0f; pose[1] = 0.0f; pose[2] = 0.0f; pose[3] = 0.0f;
            pose[4] = (cx - ppx) * davg / fx;
            pose[5] = (cy - ppy) * davg / fy;
            pose[6] = davg;
        }
    }
}

// ---------------------------------------------------------------------------
extern "C" void kernel_launch(void* const* d_in, const int* in_sizes, int n_in,
                              void* d_out, int out_size, void* d_ws, size_t ws_size,
                              hipStream_t stream)
{
    (void)in_sizes; (void)n_in; (void)out_size; (void)ws_size;

    const int*   label   = (const int*)  d_in[0];  // [1,480,640] int32
    const float* vp      = (const float*)d_in[1];  // [1,480,640,30] f32
    const float* extents = (const float*)d_in[2];  // [10,3] f32
    // d_in[3] = poses (unused by the output math)
    const float* meta    = (const float*)d_in[4];  // [1,9] f32
    float*       out     = (float*)d_out;

    // workspace layout (floats)
    float* wsf   = (float*)d_ws;
    float* u_arr = wsf;                 // 4800
    float* v_arr = wsf + P_PTS;         // 4800
    float* d_arr = wsf + 2 * P_PTS;     // 4800
    int*   lab   = (int*)(wsf + 3 * P_PTS);            // 4800
    float* hough = wsf + 4 * P_PTS;                    // 10*4800
    float* dsum  = wsf + 4 * P_PTS + NUM_CLASSES * P_PTS;

    hough_prep_kernel<<<(P_PTS + 255) / 256, 256, 0, stream>>>(
        label, vp, u_arr, v_arr, d_arr, lab);

    hough_vote_kernel<<<NTILES, 32, 0, stream>>>(
        u_arr, v_arr, d_arr, lab, hough, dsum);

    hough_finalize_kernel<<<NUM_CLASSES, 256, 0, stream>>>(
        hough, dsum, lab, extents, meta, out);
}
